// TriCNN_89515708383979
// MI455X (gfx1250) — matrix operand order repaired and checked by reading, not compiled, not run on hardware
//
#include <hip/hip_runtime.h>
#include <hip/hip_bf16.h>

// ---------------- types for WMMA (wave32, gfx1250) ----------------
typedef __attribute__((ext_vector_type(16))) __bf16 v16bf;
typedef __attribute__((ext_vector_type(8)))  __bf16 v8bf;
typedef __attribute__((ext_vector_type(8)))  float  v8f;
typedef __attribute__((ext_vector_type(4)))  float  v4f;

// ---------------- model constants ----------------
constexpr int NB   = 64;       // batch
constexpr int HH   = 11, WW = 11, DD = 11, TT = 16;
constexpr int CM   = 4;        // mid channels
constexpr int CO   = 16;       // out channels per branch
constexpr int FTOT = 516352;   // concat feature length
constexpr int N1   = 512, N2 = 256, N3 = 16;
constexpr int KT1  = FTOT / 32;   // 16136 wmma K-steps for GEMM1
constexpr int SPLITK = 128;

constexpr int COL_SP = 0;
constexpr int COL_SE = 87808;
constexpr int COL_SS = 428544;

// =====================================================================
// conv1 + bias + relu   (x:[64,1,11,11,11,16] -> a:[64,4,OH,OW,OD,16])
// =====================================================================
template<int KH,int KW,int KD,int KT,int PT,int OH,int OW,int OD>
__global__ void conv1_relu_k(const float* __restrict__ x,
                             const float* __restrict__ w,
                             const float* __restrict__ bias,
                             float* __restrict__ out) {
  constexpr int total = NB*CM*OH*OW*OD*TT;
  int idx = blockIdx.x*blockDim.x + threadIdx.x;
  if (idx >= total) return;
  int ot = idx % TT;  int t1 = idx / TT;
  int od = t1 % OD;   t1 /= OD;
  int ow = t1 % OW;   t1 /= OW;
  int oh = t1 % OH;   t1 /= OH;
  int c  = t1 % CM;   int b = t1 / CM;

  float acc = bias[c];
  #pragma unroll
  for (int kh = 0; kh < KH; ++kh)
    #pragma unroll
    for (int kw = 0; kw < KW; ++kw)
      #pragma unroll
      for (int kd = 0; kd < KD; ++kd) {
        const float* xp = x + ((((size_t)b*HH + (oh+kh))*WW + (ow+kw))*DD + (od+kd))*TT;
        const float* wp = w + ((((size_t)c*KH + kh)*KW + kw)*KD + kd)*KT;
        #pragma unroll
        for (int kt = 0; kt < KT; ++kt) {
          int it = ot + kt - PT;
          if (PT == 0 || (unsigned)it < (unsigned)TT)
            acc += xp[it] * wp[kt];
        }
      }
  out[idx] = fmaxf(acc, 0.f);
}

// =====================================================================
// per-channel BN stats (training mode): scale = g*rsqrt(var+eps),
// shift = be - mean*scale.  One block per channel (4 blocks).
// =====================================================================
__global__ void bn_stats_k(const float* __restrict__ a,
                           const float* __restrict__ g,
                           const float* __restrict__ be,
                           float* __restrict__ bnp,
                           int S /*spatial*T per sample*/, int n /*total per ch*/) {
  int c = blockIdx.x;
  __shared__ float ssum[256];
  __shared__ float ssq [256];
  float s = 0.f, q = 0.f;
  for (int i = threadIdx.x; i < n; i += blockDim.x) {
    int b = i / S, sidx = i - b*S;
    float v = a[((size_t)b*CM + c)*S + sidx];
    s += v; q += v*v;
  }
  ssum[threadIdx.x] = s; ssq[threadIdx.x] = q;
  __syncthreads();
  for (int ofs = 128; ofs > 0; ofs >>= 1) {
    if ((int)threadIdx.x < ofs) {
      ssum[threadIdx.x] += ssum[threadIdx.x + ofs];
      ssq [threadIdx.x] += ssq [threadIdx.x + ofs];
    }
    __syncthreads();
  }
  if (threadIdx.x == 0) {
    float inv_n = 1.f / (float)n;
    float mean  = ssum[0] * inv_n;
    float var   = ssq[0] * inv_n - mean*mean;
    float scale = g[c] * rsqrtf(var + 1e-3f);
    bnp[c*2+0] = scale;
    bnp[c*2+1] = be[c] - mean*scale;
  }
}

// =====================================================================
// conv2 with BN folded in + bias + relu, writes bf16 feature matrix
// a:[64,4,IH,IW,ID,16] -> fB[64, FTOT] at colBase (layout CO,OH,OW,OD,T)
// =====================================================================
template<int KH,int KW,int KD,int KT,int PT,int IH,int IW,int ID,int OH,int OW,int OD>
__global__ void conv2_bn_relu_k(const float* __restrict__ a,
                                const float* __restrict__ bnp,
                                const float* __restrict__ w,
                                const float* __restrict__ bias,
                                __bf16* __restrict__ fB, int colBase) {
  constexpr int total = NB*CO*OH*OW*OD*TT;
  int idx = blockIdx.x*blockDim.x + threadIdx.x;
  if (idx >= total) return;
  int ot = idx % TT;  int t1 = idx / TT;
  int od = t1 % OD;   t1 /= OD;
  int ow = t1 % OW;   t1 /= OW;
  int oh = t1 % OH;   t1 /= OH;
  int co = t1 % CO;   int b = t1 / CO;

  float sc0 = bnp[0], sh0 = bnp[1];
  float sc1 = bnp[2], sh1 = bnp[3];
  float sc2 = bnp[4], sh2 = bnp[5];
  float sc3 = bnp[6], sh3 = bnp[7];

  float acc = bias[co];
  #pragma unroll
  for (int ci = 0; ci < CM; ++ci) {
    float sc = (ci==0)?sc0:(ci==1)?sc1:(ci==2)?sc2:sc3;
    float sh = (ci==0)?sh0:(ci==1)?sh1:(ci==2)?sh2:sh3;
    #pragma unroll
    for (int kh = 0; kh < KH; ++kh)
      #pragma unroll
      for (int kw = 0; kw < KW; ++kw)
        #pragma unroll
        for (int kd = 0; kd < KD; ++kd) {
          const float* ap = a + ((((size_t)b*CM + ci)*IH + (oh+kh))*IW + (ow+kw))*ID*TT
                              + (size_t)(od+kd)*TT;
          const float* wp = w + (((((size_t)co*CM + ci)*KH + kh)*KW + kw)*KD + kd)*KT;
          #pragma unroll
          for (int kt = 0; kt < KT; ++kt) {
            int it = ot + kt - PT;
            if (PT == 0 || (unsigned)it < (unsigned)TT)
              acc += (ap[it]*sc + sh) * wp[kt];
          }
        }
  }
  float v = fmaxf(acc, 0.f);
  int col = colBase + ((((co*OH + oh)*OW + ow)*OD + od)*TT + ot);
  fB[(size_t)b*FTOT + col] = (__bf16)v;
}

// =====================================================================
// WMMA fragment loaders (contiguous k order, same in A and B)
// A 16x32 bf16: lanes0-15 row M=lane, K{0..15};
//               lanes16-31 same rows, K{16..31}
// B 32x16 (from row-major W[n][k], f32, converted in-register):
//   lanes0-15: N=lane, K{0..15}; lanes16-31: N=lane-16, K{16..31}
// =====================================================================
__device__ __forceinline__ v16bf load_a_bf16(const __bf16* A, long lda,
                                             int mBase, int kBase, int lane) {
  int row = mBase + (lane & 15);
  int k0  = kBase + ((lane & 16) ? 16 : 0);
  const __bf16* p = A + (long)row*lda + k0;
  v8bf c0 = *(const v8bf*)(p);        // 16B load
  v8bf c1 = *(const v8bf*)(p + 8);    // 16B load
  v16bf r;
  #pragma unroll
  for (int i = 0; i < 8; ++i) { r[i] = c0[i]; r[i+8] = c1[i]; }
  return r;
}

__device__ __forceinline__ v16bf load_b_f32(const float* Wt, long ldw,
                                            int nBase, int kBase, int lane) {
  int nrow = nBase + (lane & 15);
  int k0   = kBase + ((lane & 16) ? 16 : 0);
  const v4f* q = (const v4f*)(Wt + (long)nrow*ldw + k0);
  v4f f0 = q[0], f1 = q[1], f2 = q[2], f3 = q[3];
  v16bf r;
  #pragma unroll
  for (int i = 0; i < 4; ++i) {
    r[i]    = (__bf16)f0[i];
    r[4+i]  = (__bf16)f1[i];
    r[8+i]  = (__bf16)f2[i];
    r[12+i] = (__bf16)f3[i];
  }
  return r;
}

__device__ __forceinline__ v8f wmma_bf16(v16bf a, v16bf b, v8f c) {
  return __builtin_amdgcn_wmma_f32_16x16x32_bf16(
      false, a, false, b, (short)0, c, false, false);
}

// =====================================================================
// GEMM1: [64 x 516352](bf16) x W1[512 x 516352](f32, cvt on-fly)ᵀ
// One wave per (ks, nTile); computes ALL FOUR M-tiles so each streamed
// W1 fragment is fetched from HBM exactly once.  jobs = SPLITK*32.
// =====================================================================
__global__ void gemm1_splitk_k(const __bf16* __restrict__ A,
                               const float* __restrict__ Wt,
                               float* __restrict__ partial) {
  int lane = threadIdx.x & 31;
  // wave-uniform job index -> force into SGPR for scalar loop/addressing
  int job = __builtin_amdgcn_readfirstlane(
      (int)((blockIdx.x * blockDim.x + threadIdx.x) >> 5));
  int n  = job & 31;
  int ks = job >> 5;
  int kt0 = (ks     * KT1) / SPLITK;
  int kt1 = ((ks+1) * KT1) / SPLITK;

  v8f acc0 = {}, acc1 = {}, acc2 = {}, acc3 = {};
  const long wrow = (long)(n*16 + (lane & 15)) * FTOT + ((lane & 16) ? 16 : 0);
  for (int kt = kt0; kt < kt1; ++kt) {
    int kb = kt * 32;
    v16bf b  = load_b_f32 (Wt, FTOT, n*16, kb, lane);
    __builtin_prefetch(Wt + wrow + (kt+16)*32, 0, 1);   // speculative stream-ahead
    v16bf a0 = load_a_bf16(A, FTOT,  0, kb, lane);
    v16bf a1 = load_a_bf16(A, FTOT, 16, kb, lane);
    v16bf a2 = load_a_bf16(A, FTOT, 32, kb, lane);
    v16bf a3 = load_a_bf16(A, FTOT, 48, kb, lane);
    acc0 = wmma_bf16(a0, b, acc0);
    acc1 = wmma_bf16(a1, b, acc1);
    acc2 = wmma_bf16(a2, b, acc2);
    acc3 = wmma_bf16(a3, b, acc3);
  }
  float* base = partial + (size_t)job*1024 + lane*8;    // [job][m][lane*8..]
  *(v8f*)(base +   0) = acc0;
  *(v8f*)(base + 256) = acc1;
  *(v8f*)(base + 512) = acc2;
  *(v8f*)(base + 768) = acc3;
}

// fixed-order split-K reduction + bias + relu -> h1 (bf16 [64,512])
__global__ void reduce1_k(const float* __restrict__ partial,
                          const float* __restrict__ bias,
                          __bf16* __restrict__ h1b) {
  int e = blockIdx.x*blockDim.x + threadIdx.x;   // (32 n)*(4 m)*(256 elems)
  if (e >= 32*4*256) return;
  int t = e >> 8, r = e & 255;
  int n = t >> 2, m = t & 3;
  float s = 0.f;
  for (int ks = 0; ks < SPLITK; ++ks)
    s += partial[(size_t)((ks*32 + n)*4 + m)*256 + r];
  int lane = r >> 3, j = r & 7;
  int row = m*16 + j + ((lane & 16) ? 8 : 0);
  int col = n*16 + (lane & 15);
  float v = fmaxf(s + bias[col], 0.f);
  h1b[row*N1 + col] = (__bf16)v;
}

// =====================================================================
// Fused WMMA GEMM (no split-K): out = act(A@Wᵀ + bias). One wave/tile.
// =====================================================================
__global__ void gemm_fused_k(const __bf16* __restrict__ A, int lda,
                             const float* __restrict__ Wt, int ldw,
                             const float* __restrict__ bias,
                             __bf16* __restrict__ outB,
                             float* __restrict__ outF,
                             int Nn, int NT, int KT, int doRelu, int nJobs) {
  int lane = threadIdx.x & 31;
  int job  = __builtin_amdgcn_readfirstlane(
      (int)((blockIdx.x * blockDim.x + threadIdx.x) >> 5));
  if (job >= nJobs) return;          // wave-uniform exit (EXEC stays all-1)
  int n = job % NT, m = job / NT;
  v8f acc = {};
  for (int kt = 0; kt < KT; ++kt) {
    v16bf a = load_a_bf16(A, lda, m*16, kt*32, lane);
    v16bf b = load_b_f32 (Wt, ldw, n*16, kt*32, lane);
    acc = wmma_bf16(a, b, acc);
  }
  int col   = n*16 + (lane & 15);
  int rBase = m*16 + ((lane & 16) ? 8 : 0);
  float bcol = bias[col];
  #pragma unroll
  for (int j = 0; j < 8; ++j) {
    float v = acc[j] + bcol;
    if (doRelu) v = fmaxf(v, 0.f);
    long o = (long)(rBase + j)*Nn + col;
    outB[o] = (__bf16)v;
    if (outF) outF[o] = v;
  }
}

// =====================================================================
// softmax over 16 logits per row
// =====================================================================
__global__ void softmax_k(const float* __restrict__ logits,
                          float* __restrict__ out) {
  int r = blockIdx.x*blockDim.x + threadIdx.x;
  if (r >= NB) return;
  const float* l = logits + r*N3;
  float mx = l[0];
  #pragma unroll
  for (int j = 1; j < N3; ++j) mx = fmaxf(mx, l[j]);
  float e[N3]; float s = 0.f;
  #pragma unroll
  for (int j = 0; j < N3; ++j) { e[j] = expf(l[j]-mx); s += e[j]; }
  float inv = 1.f/s;
  #pragma unroll
  for (int j = 0; j < N3; ++j) out[r*N3 + j] = e[j]*inv;
}

// =====================================================================
// host launcher
// =====================================================================
static inline size_t align256(size_t x) { return (x + 255) & ~(size_t)255; }

extern "C" void kernel_launch(void* const* d_in, const int* in_sizes, int n_in,
                              void* d_out, int out_size, void* d_ws, size_t ws_size,
                              hipStream_t stream) {
  (void)in_sizes; (void)n_in; (void)out_size; (void)ws_size;
  const float* x     = (const float*)d_in[0];
  const float* sp_w1 = (const float*)d_in[1];
  const float* sp_b1 = (const float*)d_in[2];
  const float* sp_g  = (const float*)d_in[3];
  const float* sp_be = (const float*)d_in[4];
  const float* sp_w2 = (const float*)d_in[5];
  const float* sp_b2 = (const float*)d_in[6];
  const float* se_w1 = (const float*)d_in[7];
  const float* se_b1 = (const float*)d_in[8];
  const float* se_g  = (const float*)d_in[9];
  const float* se_be = (const float*)d_in[10];
  const float* se_w2 = (const float*)d_in[11];
  const float* se_b2 = (const float*)d_in[12];
  const float* ss_w1 = (const float*)d_in[13];
  const float* ss_b1 = (const float*)d_in[14];
  const float* ss_g  = (const float*)d_in[15];
  const float* ss_be = (const float*)d_in[16];
  const float* ss_w2 = (const float*)d_in[17];
  const float* ss_b2 = (const float*)d_in[18];
  const float* l1_w  = (const float*)d_in[19];
  const float* l1_b  = (const float*)d_in[20];
  const float* l2_w  = (const float*)d_in[21];
  const float* l2_b  = (const float*)d_in[22];
  const float* l3_w  = (const float*)d_in[23];
  const float* l3_b  = (const float*)d_in[24];
  float* out = (float*)d_out;

  // ---- workspace carve-up ----
  char* ws = (char*)d_ws; size_t off = 0;
  auto take = [&](size_t bytes) { void* p = ws + off; off = align256(off + bytes); return p; };
  float*  a_sp    = (float*) take((size_t)NB*CM*9*9*9*TT   * 4);   // 11.9 MB
  float*  a_se    = (float*) take((size_t)NB*CM*11*11*11*TT* 4);   // 21.8 MB
  float*  a_ss    = (float*) take((size_t)NB*CM*9*9*9*TT   * 4);   // 11.9 MB
  float*  bnp_sp  = (float*) take(8*4);
  float*  bnp_se  = (float*) take(8*4);
  float*  bnp_ss  = (float*) take(8*4);
  __bf16* fB      = (__bf16*)take((size_t)NB*FTOT * 2);            // 66 MB
  float*  partial = (float*) take((size_t)SPLITK*32*4*256 * 4);    // 16.8 MB
  __bf16* h1b     = (__bf16*)take((size_t)NB*N1 * 2);
  __bf16* h2b     = (__bf16*)take((size_t)NB*N2 * 2);
  float*  logitsF = (float*) take((size_t)NB*N3 * 4);
  __bf16* logitsB = (__bf16*)take((size_t)NB*N3 * 2);

  const int TB = 256;

  // ---- branch sp: k=(3,3,3,1), pad_t=0, conv1 out 9,9,9 ----
  {
    int tot1 = NB*CM*9*9*9*TT;
    conv1_relu_k<3,3,3,1,0,9,9,9><<<(tot1+TB-1)/TB, TB, 0, stream>>>(x, sp_w1, sp_b1, a_sp);
    bn_stats_k<<<CM, TB, 0, stream>>>(a_sp, sp_g, sp_be, bnp_sp, 9*9*9*TT, NB*9*9*9*TT);
    int tot2 = NB*CO*7*7*7*TT;
    conv2_bn_relu_k<3,3,3,1,0,9,9,9,7,7,7><<<(tot2+TB-1)/TB, TB, 0, stream>>>(
        a_sp, bnp_sp, sp_w2, sp_b2, fB, COL_SP);
  }
  // ---- branch se: k=(1,1,1,3), pad_t=1, conv1 out 11,11,11 ----
  {
    int tot1 = NB*CM*11*11*11*TT;
    conv1_relu_k<1,1,1,3,1,11,11,11><<<(tot1+TB-1)/TB, TB, 0, stream>>>(x, se_w1, se_b1, a_se);
    bn_stats_k<<<CM, TB, 0, stream>>>(a_se, se_g, se_be, bnp_se, 11*11*11*TT, NB*11*11*11*TT);
    int tot2 = NB*CO*11*11*11*TT;
    conv2_bn_relu_k<1,1,1,3,1,11,11,11,11,11,11><<<(tot2+TB-1)/TB, TB, 0, stream>>>(
        a_se, bnp_se, se_w2, se_b2, fB, COL_SE);
  }
  // ---- branch ss: k=(3,3,3,3), pad_t=1, conv1 out 9,9,9 ----
  {
    int tot1 = NB*CM*9*9*9*TT;
    conv1_relu_k<3,3,3,3,1,9,9,9><<<(tot1+TB-1)/TB, TB, 0, stream>>>(x, ss_w1, ss_b1, a_ss);
    bn_stats_k<<<CM, TB, 0, stream>>>(a_ss, ss_g, ss_be, bnp_ss, 9*9*9*TT, NB*9*9*9*TT);
    int tot2 = NB*CO*7*7*7*TT;
    conv2_bn_relu_k<3,3,3,3,1,9,9,9,7,7,7><<<(tot2+TB-1)/TB, TB, 0, stream>>>(
        a_ss, bnp_ss, ss_w2, ss_b2, fB, COL_SS);
  }

  // ---- GEMM1 (WMMA bf16, split-K, 4 M-tiles per wave) + reduce ----
  {
    int jobs = SPLITK * 32;                     // 4096 waves
    gemm1_splitk_k<<<jobs*32/TB, TB, 0, stream>>>(fB, l1_w, partial);
    reduce1_k<<<(32*4*256)/TB, TB, 0, stream>>>(partial, l1_b, h1b);
  }
  // ---- GEMM2: [64,512]x[256,512]ᵀ, fused relu ----
  {
    int jobs = 4 * (N2/16);                     // 64 waves
    gemm_fused_k<<<(jobs*32+TB-1)/TB, TB, 0, stream>>>(
        h1b, N1, l2_w, N1, l2_b, h2b, (float*)nullptr, N2, N2/16, N1/32, 1, jobs);
  }
  // ---- GEMM3: [64,256]x[16,256]ᵀ, no relu -> logits ----
  {
    int jobs = 4 * 1;                           // 4 waves
    gemm_fused_k<<<1, 128, 0, stream>>>(
        h2b, N2, l3_w, N2, l3_b, logitsB, logitsF, N3, 1, N2/32, 0, jobs);
  }
  // ---- softmax ----
  softmax_k<<<1, 64, 0, stream>>>(logitsF, out);
}
